// VQVAE_884763263489
// MI455X (gfx1250) — matrix operand order repaired and checked by reading, not compiled, not run on hardware
//
#include <hip/hip_runtime.h>
#include <hip/hip_bf16.h>

// ---------------------------------------------------------------------------
// VQ-VAE forward for MI455X (gfx1250, wave32).
//  - Matrix math: v_wmma_f32_16x16x32_bf16 (bf16 in, f32 accumulate)
//  - Conv weights staged to LDS by the Tensor Data Mover (tensor_load_to_lds,
//    TENSORcnt-waited, double buffered); B fragments then come from ds_load_b128.
//  - Software-pipelined A gathers (conv) and B loads (VQ argmin).
//  - Memory-bound workload: bf16 activation storage for WMMA inputs, fused
//    BN-apply+tanh, deterministic two-stage BN stats, single 4B memset.
// ---------------------------------------------------------------------------

typedef __attribute__((ext_vector_type(8)))  float  v8f;
typedef __attribute__((ext_vector_type(16))) __bf16 v16bf;
typedef unsigned int u32x4 __attribute__((ext_vector_type(4)));
typedef int          i32x8 __attribute__((ext_vector_type(8)));
typedef int          i32x4 __attribute__((ext_vector_type(4)));

union Frag {
    v16bf v;
    unsigned short u[16];
    uint4 q[2];
};

__device__ __forceinline__ unsigned short f2bf(float f) {
    unsigned u = __float_as_uint(f);
    unsigned r = u + 0x7FFFu + ((u >> 16) & 1u);       // round-to-nearest-even
    if ((u & 0x7F800000u) == 0x7F800000u) r = u;       // Inf/NaN: truncate
    return (unsigned short)(r >> 16);
}

// low 32 bits of a flat shared-memory address == LDS byte offset
__device__ __forceinline__ unsigned lds_off(const void* p) {
    return (unsigned)(uintptr_t)p;
}

// --- Tensor Data Mover: 2D panel load (rows x 64 bf16, row stride = kstride
// elements) from global into LDS. Descriptor per CDNA5 ISA section 8.3/8.4.
// clang-23 toolchain: 6-arg builtin (g0 u32x4, g1 i32x8, g2 i32x4, g3 i32x4,
// g4 i32x8, cpol i32).
__device__ __forceinline__ void tdm_load_panel(unsigned ldsaddr, const unsigned short* g,
                                               int kstride, int rows) {
    unsigned long long ga = (unsigned long long)(uintptr_t)g;
    u32x4 g0;
    g0.x = 1u;                                              // count=1, user desc
    g0.y = ldsaddr;                                         // lds_addr
    g0.z = (unsigned)ga;                                    // global_addr[31:0]
    g0.w = (unsigned)((ga >> 32) & 0x01FFFFFFu) | (2u << 30); // ga[56:32] | type=2
    const unsigned td0 = 0x40000000u, td1 = 0x40000000u;    // huge dims: no OOB clip
    i32x8 g1;
    g1[0] = (int)(1u << 16);                                // wg_mask=0, data_size=2B
    g1[1] = (int)(td0 << 16);                               // tensor_dim0[15:0] @bit48
    g1[2] = (int)((td0 >> 16) | (td1 << 16));               // dim0 hi | dim1 lo
    g1[3] = (int)((td1 >> 16) | (64u << 16));               // dim1 hi | tile_dim0=64
    g1[4] = (int)(unsigned)rows;                            // tile_dim1=rows, tile_dim2=0
    g1[5] = (int)(unsigned)kstride;                         // tensor_dim0_stride lo32
    g1[6] = 0;                                              // stride0 hi16 | stride1 lo16
    g1[7] = 0;                                              // stride1 hi32
    i32x4 g2 = {0, 0, 0, 0};                                // 2D: groups 2/3 unused
    i32x4 g3 = {0, 0, 0, 0};
    i32x8 g4 = {0, 0, 0, 0, 0, 0, 0, 0};
    __builtin_amdgcn_tensor_load_to_lds(g0, g1, g2, g3, g4, 0);
}

// --------------------------- weight / codebook prep ------------------------

// W (Co,Ci,5,5) f32  ->  wb [Npad][Kpad] bf16  (K contiguous per channel),
// k = ci*25 + (dy*5+dx), row n = co
__global__ void k_wconv(const float* __restrict__ Wf, unsigned short* __restrict__ wb,
                        int Ci, int Co, int Kpad, int Npad) {
    int i = blockIdx.x * 256 + threadIdx.x;
    int total = Npad * Kpad;
    if (i >= total) return;
    int n = i / Kpad, k = i - n * Kpad;
    int ci = k / 25, t = k - ci * 25;
    unsigned short o = 0;
    if (ci < Ci && n < Co)
        o = f2bf(Wf[((size_t)n * Ci + ci) * 25 + t]);
    wb[i] = o;
}

// codebook (8192,128) f32 -> bf16 copy + per-row squared norm
__global__ void k_cbprep(const float* __restrict__ cb, unsigned short* __restrict__ cbB,
                         float* __restrict__ cn) {
    int j = blockIdx.x;
    int t = threadIdx.x;           // 128 threads
    float c = cb[(size_t)j * 128 + t];
    cbB[(size_t)j * 128 + t] = f2bf(c);
    float s = c * c;
    #pragma unroll
    for (int m = 16; m >= 1; m >>= 1) s += __shfl_xor(s, m, 32);
    __shared__ float sh[4];
    if ((t & 31) == 0) sh[t >> 5] = s;
    __syncthreads();
    if (t == 0) cn[j] = sh[0] + sh[1] + sh[2] + sh[3];
}

// --------------------------- pad / pool / upsample / crop ------------------

__global__ void k_pad(const float* __restrict__ x, unsigned short* __restrict__ hout, int total) {
    int i = blockIdx.x * 256 + threadIdx.x;
    if (i >= total) return;
    int w = i % 256; int t = i / 256; int h = t % 256; int pc = t / 256;
    float v = 0.f;
    int hh = h - 16, ww = w - 16;
    if ((unsigned)hh < 224u && (unsigned)ww < 224u)
        v = x[((size_t)pc * 224 + hh) * 224 + ww];
    hout[i] = f2bf(v);
}

__global__ void k_pool(const float* __restrict__ fa, unsigned short* __restrict__ hout,
                       int H, int W, int total) {
    int i = blockIdx.x * 256 + threadIdx.x;
    if (i >= total) return;
    int OW = W >> 1, OH = H >> 1;
    int ow = i % OW; int t = i / OW; int oh = t % OH; int pc = t / OH;
    const float* p = fa + ((size_t)pc * H + oh * 2) * W + ow * 2;
    float m = fmaxf(fmaxf(p[0], p[1]), fmaxf(p[W], p[W + 1]));
    hout[i] = f2bf(m);
}

__global__ void k_up(const float* __restrict__ fa, unsigned short* __restrict__ hout,
                     int H, int W, int total) {
    int i = blockIdx.x * 256 + threadIdx.x;
    if (i >= total) return;
    int OW = W << 1, OH = H << 1;
    int ow = i % OW; int t = i / OW; int oh = t % OH; int pc = t / OH;
    float v = fa[((size_t)pc * H + (oh >> 1)) * W + (ow >> 1)];
    hout[i] = f2bf(v);
}

__global__ void k_crop(const float* __restrict__ fa, float* __restrict__ out) {
    int i = blockIdx.x * 256 + threadIdx.x;
    if (i >= 4 * 3 * 224 * 224) return;
    int w = i % 224; int t = i / 224; int h = t % 224; int pc = t / 224;
    out[i] = fa[((size_t)pc * 256 + h + 16) * 256 + (w + 16)];
}

// --------------------------- implicit-GEMM conv (WMMA bf16) -----------------
// One wave: 16 output pixels x (NT*16) output channels. K padded to mult of 64.
// Weights [Npad][Kpad] staged to LDS by TDM (64-K slabs, double buffered);
// A fragments im2col-gathered, software-pipelined one 32-chunk ahead.

template <int NT>
__device__ __forceinline__ void conv_gatherA(Frag& A, const unsigned short* actImg,
                                             int k0, int half, int Ci, int H, int W,
                                             int HW, int ph, int pw) {
    #pragma unroll
    for (int v = 0; v < 8; ++v) {
        #pragma unroll
        for (int j = 0; j < 2; ++j) {
            int k  = k0 + (half << 4) + (v << 1) + j;
            int ci = k / 25; int t = k - ci * 25;
            int dy = t / 5 - 2; int dx = t - (t / 5) * 5 - 2;
            int hh = ph + dy, ww = pw + dx;
            unsigned short val = 0;
            if (ci < Ci && (unsigned)hh < (unsigned)H && (unsigned)ww < (unsigned)W)
                val = actImg[(size_t)ci * HW + hh * W + ww];
            A.u[v * 2 + j] = val;
        }
    }
}

template <int NT>
__device__ __forceinline__ void conv_chunk(v8f (&c)[NT], const Frag& A,
                                           const unsigned short* bbuf, int kb,
                                           int half, int l15) {
    #pragma unroll
    for (int nt = 0; nt < NT; ++nt) {
        const unsigned short* src = bbuf + (((nt << 4) + l15) << 6) + kb + (half << 4);
        Frag B;
        B.q[0] = *(const uint4*)(src);
        B.q[1] = *(const uint4*)(src + 8);
        c[nt] = __builtin_amdgcn_wmma_f32_16x16x32_bf16(
                    false, A.v, false, B.v, (short)0, c[nt], false, false);
    }
}

template <int NT>
__global__ __launch_bounds__(256) void conv_wmma(
        const unsigned short* __restrict__ act, const unsigned short* __restrict__ wb,
        float* __restrict__ y, int Ci, int Co, int H, int W, int Kpad, int Npad) {
    __shared__ unsigned short sB[2][NT * 16 * 64];   // [buf][ncolLocal][64 k]
    const int lane = threadIdx.x & 31;
    const int wv   = threadIdx.x >> 5;
    const int mt   = blockIdx.x * 8 + wv;
    const int HW   = H * W;
    const int p0   = mt * 16;
    const int half = lane >> 4;
    const int l15  = lane & 15;
    const int cobase = blockIdx.y * (NT * 16);

    int p   = p0 + l15;
    int img = p / HW; int rr = p - img * HW;
    int ph  = rr / W; int pw = rr - ph * W;
    const unsigned short* actImg = act + (size_t)img * Ci * HW;

    const int nslab = Kpad >> 6;
    if (wv == 0)
        tdm_load_panel(lds_off(sB[0]), wb + (size_t)cobase * Kpad, Kpad, NT * 16);

    Frag A0, A1;
    conv_gatherA<NT>(A0, actImg, 0, half, Ci, H, W, HW, ph, pw);

    v8f c[NT] = {};
    for (int s = 0; s < nslab; ++s) {
        int k0 = s << 6;
        if (wv == 0) __builtin_amdgcn_s_wait_tensorcnt(0);
        __syncthreads();
        if (wv == 0 && s + 1 < nslab)
            tdm_load_panel(lds_off(sB[(s + 1) & 1]),
                           wb + (size_t)cobase * Kpad + (size_t)(s + 1) * 64,
                           Kpad, NT * 16);
        const unsigned short* bbuf = sB[s & 1];
        // chunk 0 of slab consumes A0; gather A1 for chunk 1
        conv_gatherA<NT>(A1, actImg, k0 + 32, half, Ci, H, W, HW, ph, pw);
        conv_chunk<NT>(c, A0, bbuf, 0, half, l15);
        // chunk 1 consumes A1; gather A0 for next slab (dummy re-gather at end)
        conv_gatherA<NT>(A0, actImg, (s + 1 < nslab) ? (k0 + 64) : 0,
                         half, Ci, H, W, HW, ph, pw);
        conv_chunk<NT>(c, A1, bbuf, 32, half, l15);
        __syncthreads();
    }

    // C/D layout: lane L holds (m = v + (L>=16 ? 8 : 0), n = L&15) in VGPR v
    #pragma unroll
    for (int nt = 0; nt < NT; ++nt) {
        int co = cobase + nt * 16 + l15;
        if (co >= Co) continue;
        #pragma unroll
        for (int v = 0; v < 8; ++v) {
            int pp  = p0 + v + half * 8;
            int im2 = pp / HW; int r2 = pp - im2 * HW;
            y[((size_t)im2 * Co + co) * HW + r2] = c[nt][v];
        }
    }
}

// --------------------------- BN stats + apply + tanh ------------------------

__global__ void k_stats_part(const float* __restrict__ y, float* __restrict__ part,
                             int Co, int HW) {
    int c = blockIdx.x, s = blockIdx.y;
    int tid = threadIdx.x;
    int total = 4 * HW;
    float s1 = 0.f, s2 = 0.f;
    for (int e = s * 256 + tid; e < total; e += 16 * 256) {
        int img = e / HW; int i = e - img * HW;
        float v = y[((size_t)img * Co + c) * HW + i];
        s1 += v; s2 += v * v;
    }
    __shared__ float a1[256], a2[256];
    a1[tid] = s1; a2[tid] = s2; __syncthreads();
    for (int st = 128; st > 0; st >>= 1) {
        if (tid < st) { a1[tid] += a1[tid + st]; a2[tid] += a2[tid + st]; }
        __syncthreads();
    }
    if (tid == 0) { part[(c * 16 + s) * 2] = a1[0]; part[(c * 16 + s) * 2 + 1] = a2[0]; }
}

__global__ void k_stats_final(const float* __restrict__ part, const float* __restrict__ gamma,
                              const float* __restrict__ beta, float* __restrict__ ss,
                              int Co, int n) {
    int c = blockIdx.x;
    if (threadIdx.x != 0) return;
    float s1 = 0.f, s2 = 0.f;
    for (int s = 0; s < 16; ++s) { s1 += part[(c * 16 + s) * 2]; s2 += part[(c * 16 + s) * 2 + 1]; }
    float m = s1 / (float)n;
    float v = s2 / (float)n - m * m;
    if (v < 0.f) v = 0.f;
    float sc = gamma[c] * rsqrtf(v + 1e-5f);
    ss[c]      = sc;
    ss[Co + c] = beta[c] - m * sc;
}

__global__ void k_bn_tanh(const float* __restrict__ y, const float* __restrict__ ss,
                          float* __restrict__ fa, unsigned short* __restrict__ hb,
                          int Co, int HW, int total) {
    int i = blockIdx.x * 256 + threadIdx.x;
    if (i >= total) return;
    int c = (i / HW) % Co;
    float t = tanhf(fmaf(y[i], ss[c], ss[Co + c]));
    fa[i] = t;
    hb[i] = f2bf(t);
}

// --------------------------- VQ: argmin via WMMA ----------------------------
// d2 = ||f||^2 - 2 f.c + ||c||^2 ; row-norm term constant per row -> dropped.
// B loads double-buffered one column-tile ahead of the WMMAs.

__device__ __forceinline__ void vq_loadB(Frag (&B)[4], const unsigned short* cbB,
                                         int jt, int l15, int half) {
    int col = jt * 16 + l15;
    #pragma unroll
    for (int kk = 0; kk < 4; ++kk) {
        const unsigned short* src = cbB + ((size_t)col << 7) + (kk << 5) + (half << 4);
        B[kk].q[0] = *(const uint4*)(src);
        B[kk].q[1] = *(const uint4*)(src + 8);
    }
}

__device__ __forceinline__ void vq_step(const Frag (&A)[4], const Frag (&B)[4],
                                        const float* cn, int jt, int l15,
                                        float (&bestd)[8], int (&besti)[8]) {
    v8f c = {};
    #pragma unroll
    for (int kk = 0; kk < 4; ++kk)
        c = __builtin_amdgcn_wmma_f32_16x16x32_bf16(
                false, A[kk].v, false, B[kk].v, (short)0, c, false, false);
    int col = jt * 16 + l15;
    float cno = cn[col];
    #pragma unroll
    for (int v = 0; v < 8; ++v) {
        float d2 = cno - 2.0f * c[v];
        if (d2 < bestd[v]) { bestd[v] = d2; besti[v] = col; }
    }
}

__global__ __launch_bounds__(256) void k_vq_argmin(
        const unsigned short* __restrict__ flatB, const unsigned short* __restrict__ cbB,
        const float* __restrict__ cn, int* __restrict__ idx) {
    const int lane = threadIdx.x & 31;
    const int wv   = threadIdx.x >> 5;
    const int tile = blockIdx.x * 8 + wv;     // 256 tiles of 16 rows
    const int half = lane >> 4;
    const int l15  = lane & 15;
    const int row  = tile * 16 + l15;

    Frag A[4];
    #pragma unroll
    for (int kk = 0; kk < 4; ++kk) {
        const unsigned short* src = flatB + ((size_t)row << 7) + (kk << 5) + (half << 4);
        A[kk].q[0] = *(const uint4*)(src);
        A[kk].q[1] = *(const uint4*)(src + 8);
    }

    float bestd[8]; int besti[8];
    #pragma unroll
    for (int v = 0; v < 8; ++v) { bestd[v] = 3.4e38f; besti[v] = 0; }

    Frag Ba[4], Bb[4];
    vq_loadB(Ba, cbB, 0, l15, half);
    for (int jt = 0; jt < 512; jt += 2) {
        vq_loadB(Bb, cbB, jt + 1, l15, half);
        vq_step(A, Ba, cn, jt, l15, bestd, besti);
        if (jt + 2 < 512) vq_loadB(Ba, cbB, jt + 2, l15, half);
        vq_step(A, Bb, cn, jt + 1, l15, bestd, besti);
    }

    #pragma unroll
    for (int m = 8; m >= 1; m >>= 1) {
        #pragma unroll
        for (int v = 0; v < 8; ++v) {
            float od = __shfl_xor(bestd[v], m, 32);
            int   oi = __shfl_xor(besti[v], m, 32);
            if (od < bestd[v] || (od == bestd[v] && oi < besti[v])) { bestd[v] = od; besti[v] = oi; }
        }
    }
    if (l15 == 0) {
        #pragma unroll
        for (int v = 0; v < 8; ++v)
            idx[tile * 16 + half * 8 + v] = besti[v];
    }
}

__global__ void k_vq_gather(const float* __restrict__ cb, const int* __restrict__ idx,
                            const float* __restrict__ latF, unsigned short* __restrict__ xqB,
                            float* __restrict__ loss) {
    int e = blockIdx.x * 256 + threadIdx.x;     // total 4096*128
    int row = e >> 7, k = e & 127;
    float c = cb[(size_t)idx[row] * 128 + k];
    xqB[e] = f2bf(c);
    float d = c - latF[e];
    __shared__ float sh[256];
    sh[threadIdx.x] = d * d; __syncthreads();
    for (int st = 128; st > 0; st >>= 1) {
        if (threadIdx.x < st) sh[threadIdx.x] += sh[threadIdx.x + st];
        __syncthreads();
    }
    if (threadIdx.x == 0) atomicAdd(loss, sh[0]);
}

__global__ void k_loss_final(const float* __restrict__ loss, float* __restrict__ out) {
    // codebook_loss + commitment_loss = 2 * mean((xq-x)^2)
    out[4 * 3 * 224 * 224] = 2.0f * loss[0] / (4096.0f * 128.0f);
}

// --------------------------- host orchestration -----------------------------

extern "C" void kernel_launch(void* const* d_in, const int* in_sizes, int n_in,
                              void* d_out, int out_size, void* d_ws, size_t ws_size,
                              hipStream_t stream) {
    (void)in_sizes; (void)n_in; (void)out_size; (void)ws_size;
    const float* x        = (const float*)d_in[0];
    const float* codebook = (const float*)d_in[65];

    static const int LCi[16] = {3,32, 32,64, 64,128, 128,128, 128,128, 64,64, 32,32, 3,3};
    static const int LCo[16] = {32,32, 64,64, 128,128, 128,128, 128,64, 64,32, 32,3, 3,3};
    static const int LHW[16] = {256,256, 128,128, 64,64, 32,32, 32,32, 64,64, 128,128, 256,256};
    // pool after L1,L3,L5 ; VQ after L7 ; upsample after L9,L11,L13 ; crop after L15.

    char* ws = (char*)d_ws;
    size_t off = 0;
    auto alloc = [&](size_t bytes) -> char* {
        char* p = ws + off;
        off = (off + bytes + 255) & ~(size_t)255;
        return p;
    };
    unsigned short* HBUF0 = (unsigned short*)alloc(8388608ull * 2);
    unsigned short* HBUF1 = (unsigned short*)alloc(8388608ull * 2);
    float* FY = (float*)alloc(8388608ull * 4);
    float* FA = (float*)alloc(8388608ull * 4);

    unsigned short* WBL[16];
    int Kpad[16], Npad[16], NTv[16];
    for (int L = 0; L < 16; ++L) {
        int Ci = LCi[L], Co = LCo[L];
        Kpad[L] = ((Ci * 25 + 63) / 64) * 64;   // multiple of 64 (two WMMA chunks/slab)
        int nt  = (Co >= 64) ? 4 : ((Co >= 32) ? 2 : 1);
        NTv[L]  = nt;
        int g   = nt * 16;
        Npad[L] = ((Co + g - 1) / g) * g;
        WBL[L]  = (unsigned short*)alloc((size_t)Kpad[L] * Npad[L] * 2);
    }
    unsigned short* CBB  = (unsigned short*)alloc(8192ull * 128 * 2);
    float* CN   = (float*)alloc(8192ull * 4);
    int*   IDX  = (int*)alloc(4096ull * 4);
    float* PART = (float*)alloc(128 * 16 * 2 * 4);
    float* SS   = (float*)alloc(2 * 128 * 4);
    float* LOSS = (float*)alloc(256);

    for (int L = 0; L < 16; ++L) {
        const float* Wf = (const float*)d_in[1 + L * 4];
        int total = Npad[L] * Kpad[L];
        k_wconv<<<(total + 255) / 256, 256, 0, stream>>>(Wf, WBL[L], LCi[L], LCo[L], Kpad[L], Npad[L]);
    }
    k_cbprep<<<8192, 128, 0, stream>>>(codebook, CBB, CN);
    (void)hipMemsetAsync(LOSS, 0, sizeof(float), stream);

    {
        int total = 4 * 3 * 256 * 256;
        k_pad<<<(total + 255) / 256, 256, 0, stream>>>(x, HBUF0, total);
    }

    unsigned short* hin  = HBUF0;
    unsigned short* hout = HBUF1;

    for (int L = 0; L < 16; ++L) {
        int Ci = LCi[L], Co = LCo[L], Hs = LHW[L], Wsp = LHW[L], HW = Hs * Wsp;
        int Mtiles = (4 * HW) / 16;
        dim3 grid(Mtiles / 8, Npad[L] / (NTv[L] * 16));
        switch (NTv[L]) {
            case 4: conv_wmma<4><<<grid, 256, 0, stream>>>(hin, WBL[L], FY, Ci, Co, Hs, Wsp, Kpad[L], Npad[L]); break;
            case 2: conv_wmma<2><<<grid, 256, 0, stream>>>(hin, WBL[L], FY, Ci, Co, Hs, Wsp, Kpad[L], Npad[L]); break;
            default: conv_wmma<1><<<grid, 256, 0, stream>>>(hin, WBL[L], FY, Ci, Co, Hs, Wsp, Kpad[L], Npad[L]); break;
        }
        k_stats_part<<<dim3(Co, 16), 256, 0, stream>>>(FY, PART, Co, HW);
        k_stats_final<<<Co, 32, 0, stream>>>(PART, (const float*)d_in[1 + L * 4 + 2],
                                             (const float*)d_in[1 + L * 4 + 3], SS, Co, 4 * HW);
        int total = 4 * Co * HW;
        k_bn_tanh<<<(total + 255) / 256, 256, 0, stream>>>(FY, SS, FA, hout, Co, HW, total);

        if (L == 1 || L == 3 || L == 5) {
            int ot = 4 * Co * (HW / 4);
            k_pool<<<(ot + 255) / 256, 256, 0, stream>>>(FA, hin, Hs, Wsp, ot);
        } else if (L == 9 || L == 11 || L == 13) {
            int ot = 4 * Co * HW * 4;
            k_up<<<(ot + 255) / 256, 256, 0, stream>>>(FA, hin, Hs, Wsp, ot);
        } else if (L == 7) {
            k_vq_argmin<<<32, 256, 0, stream>>>(hout, CBB, CN, IDX);
            k_vq_gather<<<2048, 256, 0, stream>>>(codebook, IDX, FA, hin, LOSS);
        } else {
            unsigned short* t = hin; hin = hout; hout = t;
        }
    }

    k_crop<<<(4 * 3 * 224 * 224 + 255) / 256, 256, 0, stream>>>(FA, (float*)d_out);
    k_loss_final<<<1, 1, 0, stream>>>(LOSS, (float*)d_out);
}
